// TransformerBlock_21844203668377
// MI455X (gfx1250) — matrix operand order repaired and checked by reading, not compiled, not run on hardware
//
#include <hip/hip_runtime.h>
#include <hip/hip_bf16.h>

#define N_TOK 8192
#define F_DIM 16
#define H_HEADS 4
#define D_HEAD 4
#define KQE 32
#define HID 512
#define EPSV 1.1920929e-07f

typedef _Float16 h8  __attribute__((ext_vector_type(8)));
typedef _Float16 v16h __attribute__((ext_vector_type(16)));
typedef float    v8f __attribute__((ext_vector_type(8)));

static __device__ __forceinline__ v16h mk16(h8 lo, h8 hi) {
  v16h r;
#pragma unroll
  for (int i = 0; i < 8; i++) { r[i] = lo[i]; r[i + 8] = hi[i]; }
  return r;
}
static __device__ __forceinline__ v8f zero8() {
  v8f c;
#pragma unroll
  for (int i = 0; i < 8; i++) c[i] = 0.0f;
  return c;
}
static __device__ __forceinline__ float silu_f(float v) {
  return v / (1.0f + __expf(-v));
}

// ---- DPP-based xor "shuffles" within a 16-lane row: pure VALU, no LDS/waits.
template <int CTRL>
static __device__ __forceinline__ float dpp_xor(float v) {
  int i = __builtin_bit_cast(int, v);
  int r = __builtin_amdgcn_update_dpp(i, i, CTRL, 0xf, 0xf, true);
  return __builtin_bit_cast(float, r);
}
template <int CTRL, int N>
static __device__ __forceinline__ void step_max(float* a) {
  float t[N];
#pragma unroll
  for (int r = 0; r < N; r++) t[r] = dpp_xor<CTRL>(a[r]);
#pragma unroll
  for (int r = 0; r < N; r++) a[r] = fmaxf(a[r], t[r]);
}
template <int CTRL, int N>
static __device__ __forceinline__ void step_sum(float* a) {
  float t[N];
#pragma unroll
  for (int r = 0; r < N; r++) t[r] = dpp_xor<CTRL>(a[r]);
#pragma unroll
  for (int r = 0; r < N; r++) a[r] += t[r];
}
template <int N>
static __device__ __forceinline__ void rowmax16(float* a) {
  step_max<0xB1, N>(a); step_max<0x4E, N>(a);
  step_max<0x164, N>(a); step_max<0x168, N>(a);
}
template <int N>
static __device__ __forceinline__ void rowsum16(float* a) {
  step_sum<0xB1, N>(a); step_sum<0x4E, N>(a);
  step_sum<0x164, N>(a); step_sum<0x168, N>(a);
}

// ---------------------------------------------------------------------------
// Kernel 0: convert MLP weights to f16
// ---------------------------------------------------------------------------
__global__ void prep_weights(const float* __restrict__ w1,
                             const float* __restrict__ w0,
                             const float* __restrict__ w2,
                             _Float16* __restrict__ w1h,
                             _Float16* __restrict__ w0h,
                             _Float16* __restrict__ w2h) {
  int i = blockIdx.x * blockDim.x + threadIdx.x;
  int stride = gridDim.x * blockDim.x;
  for (int k = i; k < HID * HID; k += stride) w1h[k] = (_Float16)w1[k];
  for (int k = i; k < HID * F_DIM; k += stride) {
    w0h[k] = (_Float16)w0[k];
    w2h[k] = (_Float16)w2[k];
  }
}

// ---------------------------------------------------------------------------
// Kernel 1: RMSNorm1 + Q/K/V projections. One thread per token row.
// Q is pre-scaled by 1/sqrt(KQ) so the attention loop needs no scaling.
// ---------------------------------------------------------------------------
__global__ void qkv_kernel(const float* __restrict__ x, const float* __restrict__ n1w,
                           const float* __restrict__ Qw, const float* __restrict__ Qb,
                           const float* __restrict__ Kw, const float* __restrict__ Kb,
                           const float* __restrict__ Vw, const float* __restrict__ Vb,
                           _Float16* __restrict__ Qh, _Float16* __restrict__ Kh,
                           float* __restrict__ Vbuf) {
  int n = blockIdx.x * blockDim.x + threadIdx.x;
  if (n >= N_TOK) return;
  const float qscale = 0.17677669529663687f;  // 1/sqrt(32)
  float xr[16], xn[16];
  float ss = 0.f;
#pragma unroll
  for (int f = 0; f < 16; f++) { xr[f] = x[(size_t)n * 16 + f]; ss += xr[f] * xr[f]; }
  float rn = rsqrtf(ss * (1.0f / 16.0f) + EPSV);
#pragma unroll
  for (int f = 0; f < 16; f++) xn[f] = xr[f] * rn * n1w[f];

  for (int h = 0; h < H_HEADS; h++) {
    for (int e = 0; e < KQE; e++) {
      float aq = Qb[h * KQE + e];
      float ak = Kb[h * KQE + e];
#pragma unroll
      for (int f = 0; f < 16; f++) {
        aq += xn[f] * Qw[(h * KQE + e) * 16 + f];
        ak += xn[f] * Kw[(h * KQE + e) * 16 + f];
      }
      size_t o = ((size_t)h * N_TOK + n) * KQE + e;
      Qh[o] = (_Float16)(aq * qscale);
      Kh[o] = (_Float16)ak;
    }
#pragma unroll
    for (int d = 0; d < D_HEAD; d++) {
      float av = Vb[h * D_HEAD + d];
#pragma unroll
      for (int f = 0; f < 16; f++) av += xn[f] * Vw[(h * D_HEAD + d) * 16 + f];
      Vbuf[((size_t)h * N_TOK + n) * D_HEAD + d] = av;
    }
  }
}

// ---------------------------------------------------------------------------
// Kernel 2: flash attention, one wave per (head, 16-query tile).
// 4 key tiles (64 keys) per softmax round; DPP row reductions; Q pre-scaled.
// ---------------------------------------------------------------------------
__global__ void __launch_bounds__(256, 1) attn_kernel(
    const _Float16* __restrict__ Qh,
    const _Float16* __restrict__ Kh,
    const float* __restrict__ Vbuf,
    float* __restrict__ results) {
  const int lane = threadIdx.x & 31;
  const int wid = blockIdx.x * (blockDim.x >> 5) + (threadIdx.x >> 5);
  const int h = wid & 3;
  const int m0 = (wid >> 2) << 4;
  const int mrow = lane & 15;
  const int hi = lane >> 4;

  // A fragment: k = j + 16*hi
  const _Float16* qp = Qh + ((size_t)h * N_TOK + m0 + mrow) * KQE + hi * 16;
  v16h aQ = mk16(*(const h8*)qp, *(const h8*)(qp + 8));

  // per-lane base pointers for K (B fragment: k = j + 16*hi) and V
  const _Float16* kbase = Kh + ((size_t)h * N_TOK + mrow) * KQE + hi * 16;
  const float* vbase = Vbuf + ((size_t)h * N_TOK + mrow) * 4;

  float mstate[8], lstate[8], o[8][4];
#pragma unroll
  for (int r = 0; r < 8; r++) {
    mstate[r] = -3.0e38f;
    lstate[r] = 0.f;
#pragma unroll
    for (int d = 0; d < 4; d++) o[r][d] = 0.f;
  }

  for (int t = 0; t < N_TOK / 64; t++) {
    const int kb = t * 64;
    const _Float16* kp = kbase + (size_t)kb * KQE;
    v16h bK0 = *(const v16h*)(kp);
    v16h bK1 = *(const v16h*)(kp + 16 * KQE);
    v16h bK2 = *(const v16h*)(kp + 32 * KQE);
    v16h bK3 = *(const v16h*)(kp + 48 * KQE);
    if (t + 1 < N_TOK / 64) __builtin_prefetch(kp + 64 * KQE, 0, 3);

    v8f s0 = __builtin_amdgcn_wmma_f32_16x16x32_f16(
        false, aQ, false, bK0, (short)0, zero8(), false, false);
    v8f s1 = __builtin_amdgcn_wmma_f32_16x16x32_f16(
        false, aQ, false, bK1, (short)0, zero8(), false, false);
    v8f s2 = __builtin_amdgcn_wmma_f32_16x16x32_f16(
        false, aQ, false, bK2, (short)0, zero8(), false, false);
    v8f s3 = __builtin_amdgcn_wmma_f32_16x16x32_f16(
        false, aQ, false, bK3, (short)0, zero8(), false, false);

    float4 vv0 = *(const float4*)(vbase + (size_t)(kb)*4);
    float4 vv1 = *(const float4*)(vbase + (size_t)(kb + 16) * 4);
    float4 vv2 = *(const float4*)(vbase + (size_t)(kb + 32) * 4);
    float4 vv3 = *(const float4*)(vbase + (size_t)(kb + 48) * 4);
    float v0d[4] = {vv0.x, vv0.y, vv0.z, vv0.w};
    float v1d[4] = {vv1.x, vv1.y, vv1.z, vv1.w};
    float v2d[4] = {vv2.x, vv2.y, vv2.z, vv2.w};
    float v3d[4] = {vv3.x, vv3.y, vv3.z, vv3.w};

    float p0[8], p1[8], p2[8], p3[8], rmax[8];
#pragma unroll
    for (int r = 0; r < 8; r++) {
      p0[r] = s0[r];
      p1[r] = s1[r];
      p2[r] = s2[r];
      p3[r] = s3[r];
      rmax[r] = fmaxf(fmaxf(p0[r], p1[r]), fmaxf(p2[r], p3[r]));
    }
    rowmax16<8>(rmax);

    float rsum[8], alpha[8];
#pragma unroll
    for (int r = 0; r < 8; r++) {
      float mn = fmaxf(mstate[r], rmax[r]);
      alpha[r] = __expf(mstate[r] - mn);
      mstate[r] = mn;
      p0[r] = __expf(p0[r] - mn);
      p1[r] = __expf(p1[r] - mn);
      p2[r] = __expf(p2[r] - mn);
      p3[r] = __expf(p3[r] - mn);
      rsum[r] = (p0[r] + p1[r]) + (p2[r] + p3[r]);
    }
    rowsum16<8>(rsum);

#pragma unroll
    for (int r = 0; r < 8; r++) {
      lstate[r] = lstate[r] * alpha[r] + rsum[r];
#pragma unroll
      for (int d = 0; d < 4; d++) {
        float acc = o[r][d] * alpha[r];
        acc += p0[r] * v0d[d];
        acc += p1[r] * v1d[d];
        acc += p2[r] * v2d[d];
        acc += p3[r] * v3d[d];
        o[r][d] = acc;
      }
    }
  }

  // reduce O partials across the 16 key lanes (flattened 32 floats)
  float of[32];
#pragma unroll
  for (int r = 0; r < 8; r++)
#pragma unroll
    for (int d = 0; d < 4; d++) of[r * 4 + d] = o[r][d];
  rowsum16<32>(of);

#pragma unroll
  for (int r = 0; r < 8; r++) {
    float inv = 1.0f / lstate[r];
    int row = m0 + r + 8 * hi;
#pragma unroll
    for (int d = 0; d < 4; d++) {
      if (mrow == d)
        results[(size_t)row * 16 + h * 4 + d] = of[r * 4 + d] * inv;
    }
  }
}

// ---------------------------------------------------------------------------
// Kernel 3: out-proj + residual + RMSNorm2 + 3-layer SiLU MLP (WMMA) + residual
// ---------------------------------------------------------------------------
__global__ void __launch_bounds__(32) mlp_kernel(
    const float* __restrict__ x, const float* __restrict__ results,
    const float* __restrict__ attn_w, const float* __restrict__ attn_b,
    const float* __restrict__ n2w,
    const float* __restrict__ b0, const float* __restrict__ b1,
    const float* __restrict__ b2,
    const _Float16* __restrict__ w0h, const _Float16* __restrict__ w1h,
    const _Float16* __restrict__ w2h, float* __restrict__ out) {
  __shared__ float zsh[16][16];
  __shared__ _Float16 znh[16][16];
  __shared__ _Float16 h0sh[16][HID];
  __shared__ _Float16 h1sh[16][HID];

  const int lane = threadIdx.x & 31;
  const int mrow = lane & 15;
  const int hi = lane >> 4;
  const int m0 = blockIdx.x * 16;
  const int row = m0 + mrow;

  // ---- out-projection + residual + RMSNorm2
  float res[16];
#pragma unroll
  for (int g = 0; g < 16; g++) res[g] = results[(size_t)row * 16 + g];
  float zr[8];
  float ss = 0.f;
  const int f0 = hi * 8;
#pragma unroll
  for (int j = 0; j < 8; j++) {
    int f = f0 + j;
    float acc = attn_b[f];
#pragma unroll
    for (int g = 0; g < 16; g++) acc += res[g] * attn_w[f * 16 + g];
    float z = x[(size_t)row * 16 + f] + acc;
    zr[j] = z;
    ss += z * z;
    zsh[mrow][f] = z;
  }
  ss += __shfl_xor(ss, 16, 32);  // cross half-wave: once per wave
  float rn = rsqrtf(ss * (1.0f / 16.0f) + EPSV);
#pragma unroll
  for (int j = 0; j < 8; j++) {
    int f = f0 + j;
    znh[mrow][f] = (_Float16)(zr[j] * rn * n2w[f]);
  }
  __syncthreads();

  // ---- layer 1: h0 = silu(zn @ w0.T + b0), K=16 zero-padded to 32
  h8 alo = *(const h8*)&znh[mrow][hi * 8];
  v16h aZ;
#pragma unroll
  for (int i = 0; i < 8; i++) { aZ[i] = alo[i]; aZ[i + 8] = (_Float16)0.f; }
  for (int ct = 0; ct < HID / 16; ct++) {
    int ncol = ct * 16 + mrow;
    v16h bW;
#pragma unroll
    for (int i = 0; i < 16; i++) bW[i] = (_Float16)0.f;
    { h8 wl = *(const h8*)(w0h + (size_t)ncol * 16 + hi * 8); for (int i = 0; i < 8; i++) bW[i] = wl[i]; }  // k=8*hi..8*hi+7
    v8f acc = __builtin_amdgcn_wmma_f32_16x16x32_f16(
        false, aZ, false, bW, (short)0, zero8(), false, false);
    float bb = b0[ncol];
#pragma unroll
    for (int r = 0; r < 8; r++)
      h0sh[r + 8 * hi][ncol] = (_Float16)silu_f(acc[r] + bb);
  }
  __syncthreads();

  // ---- layer 2: h1 = silu(h0 @ w1.T + b1)  (the big 512x512 GEMM)
  for (int ct = 0; ct < HID / 16; ct++) {
    int ncol = ct * 16 + mrow;
    v8f acc = zero8();
    for (int kt = 0; kt < HID / 32; kt++) {
      const _Float16* ap = &h0sh[mrow][kt * 32 + hi * 16];
      v16h a = mk16(*(const h8*)ap, *(const h8*)(ap + 8));
      const _Float16* bp = w1h + (size_t)ncol * HID + kt * 32 + hi * 16;
      v16h b = *(const v16h*)bp;
      acc = __builtin_amdgcn_wmma_f32_16x16x32_f16(
          false, a, false, b, (short)0, acc, false, false);
    }
    float bb = b1[ncol];
#pragma unroll
    for (int r = 0; r < 8; r++)
      h1sh[r + 8 * hi][ncol] = (_Float16)silu_f(acc[r] + bb);
  }
  __syncthreads();

  // ---- layer 3: out = silu(h1 @ w2.T + b2) + z
  {
    v8f acc = zero8();
    for (int kt = 0; kt < HID / 32; kt++) {
      const _Float16* ap = &h1sh[mrow][kt * 32 + hi * 16];
      v16h a = mk16(*(const h8*)ap, *(const h8*)(ap + 8));
      const _Float16* bp = w2h + (size_t)mrow * HID + kt * 32 + hi * 16;
      v16h b = *(const v16h*)bp;
      acc = __builtin_amdgcn_wmma_f32_16x16x32_f16(
          false, a, false, b, (short)0, acc, false, false);
    }
    float bb = b2[mrow];
#pragma unroll
    for (int r = 0; r < 8; r++) {
      int m = r + 8 * hi;
      out[(size_t)(m0 + m) * 16 + mrow] = silu_f(acc[r] + bb) + zsh[m][mrow];
    }
  }
}

// ---------------------------------------------------------------------------
extern "C" void kernel_launch(void* const* d_in, const int* in_sizes, int n_in,
                              void* d_out, int out_size, void* d_ws, size_t ws_size,
                              hipStream_t stream) {
  (void)in_sizes; (void)n_in; (void)out_size; (void)ws_size;
  const float* x      = (const float*)d_in[0];
  const float* n1w    = (const float*)d_in[1];
  const float* n2w    = (const float*)d_in[2];
  const float* Qw     = (const float*)d_in[3];
  const float* Qb     = (const float*)d_in[4];
  const float* Kw     = (const float*)d_in[5];
  const float* Kb     = (const float*)d_in[6];
  const float* Vw     = (const float*)d_in[7];
  const float* Vb     = (const float*)d_in[8];
  const float* attn_w = (const float*)d_in[9];
  const float* attn_b = (const float*)d_in[10];
  const float* w0     = (const float*)d_in[11];
  const float* b0     = (const float*)d_in[12];
  const float* w1     = (const float*)d_in[13];
  const float* b1     = (const float*)d_in[14];
  const float* w2     = (const float*)d_in[15];
  const float* b2     = (const float*)d_in[16];
  float* out = (float*)d_out;

  char* ws = (char*)d_ws;
  size_t off = 0;
  auto carve = [&](size_t bytes) -> char* {
    char* p = ws + off;
    off += (bytes + 255) & ~(size_t)255;
    return p;
  };
  _Float16* Qh   = (_Float16*)carve((size_t)H_HEADS * N_TOK * KQE * 2);
  _Float16* Kh   = (_Float16*)carve((size_t)H_HEADS * N_TOK * KQE * 2);
  float*    Vbuf = (float*)carve((size_t)H_HEADS * N_TOK * D_HEAD * 4);
  float*    resb = (float*)carve((size_t)N_TOK * F_DIM * 4);
  _Float16* w1h  = (_Float16*)carve((size_t)HID * HID * 2);
  _Float16* w0h  = (_Float16*)carve((size_t)HID * F_DIM * 2);
  _Float16* w2h  = (_Float16*)carve((size_t)HID * F_DIM * 2);

  prep_weights<<<256, 256, 0, stream>>>(w1, w0, w2, w1h, w0h, w2h);
  qkv_kernel<<<N_TOK / 256, 256, 0, stream>>>(x, n1w, Qw, Qb, Kw, Kb, Vw, Vb,
                                              Qh, Kh, Vbuf);
  attn_kernel<<<(H_HEADS * (N_TOK / 16)) / 8, 256, 0, stream>>>(Qh, Kh, Vbuf, resb);
  mlp_kernel<<<N_TOK / 16, 32, 0, stream>>>(x, resb, attn_w, attn_b, n2w,
                                            b0, b1, b2, w0h, w1h, w2h, out);
}
